// CSWin_8375186227707
// MI455X (gfx1250) — matrix operand order repaired and checked by reading, not compiled, not run on hardware
//
#include <hip/hip_runtime.h>

typedef __attribute__((ext_vector_type(16))) __bf16 v16bf;
typedef __attribute__((ext_vector_type(8)))  float  v8f;

#define CH    128   // keys staged per chunk
#define KSTR  40    // ushort stride of K [token][dim] bf16 tile (80B rows, 16B aligned)
#define VSTR  136   // ushort stride of V [dim][token] bf16 chunk (272B rows, 16B aligned)
#define PSTR  72    // ushort stride of per-wave P tile rows (144B, 16B aligned)

#if __has_builtin(__builtin_amdgcn_exp2f)
#define EXP2(x) __builtin_amdgcn_exp2f(x)
#else
#define EXP2(x) exp2f(x)
#endif
#if __has_builtin(__builtin_amdgcn_rcpf)
#define RCP(x) __builtin_amdgcn_rcpf(x)
#else
#define RCP(x) (1.0f / (x))
#endif

// truncating bf16 (compiler emits ds_store_b16_d16_hi for LDS stores)
__device__ __forceinline__ unsigned short t2bf(float f) {
  return (unsigned short)(__float_as_uint(f) >> 16);
}
// truncating pack of two f32 -> {bf16(a), bf16(b)} in one u32 (v_and_or_b32)
__device__ __forceinline__ unsigned int pk2bf(float a, float b) {
  return (__float_as_uint(a) >> 16) | (__float_as_uint(b) & 0xFFFF0000u);
}

union Frag { unsigned int u[8]; uint4 q[2]; v16bf v; };

// Fused DPP butterfly steps: one VOP2+DPP16 instruction computes
// d = op(dpp_perm(x), x). 16-lane rows match the WMMA C-layout halves.
#define DPP_MAX_STEP(d, x, CTRL)                                              \
  asm("v_max_num_f32_dpp %0, %1, %1 " CTRL                                    \
      " row_mask:0xf bank_mask:0xf bound_ctrl:1"                              \
      : "=&v"(d) : "v"(x))
#define DPP_ADD_STEP(d, x, CTRL)                                              \
  asm("v_add_f32_dpp %0, %1, %1 " CTRL                                        \
      " row_mask:0xf bank_mask:0xf bound_ctrl:1"                              \
      : "=&v"(d) : "v"(x))

__device__ __forceinline__ float red16_max(float x) {
  float t;
  DPP_MAX_STEP(t, x, "quad_perm:[1,0,3,2]"); x = t;   // xor1
  DPP_MAX_STEP(t, x, "quad_perm:[2,3,0,1]"); x = t;   // xor2
  DPP_MAX_STEP(t, x, "row_half_mirror");     x = t;   // xor4 (groups uniform)
  DPP_MAX_STEP(t, x, "row_mirror");          x = t;   // xor8
  return x;
}
__device__ __forceinline__ float red16_sum(float x) {
  float t;
  DPP_ADD_STEP(t, x, "quad_perm:[1,0,3,2]"); x = t;
  DPP_ADD_STEP(t, x, "quad_perm:[2,3,0,1]"); x = t;
  DPP_ADD_STEP(t, x, "row_half_mirror");     x = t;
  DPP_ADD_STEP(t, x, "row_mirror");          x = t;
  return x;
}

// ---------------------------------------------------------------------------
// rpe = depthwise 3x3 conv over per-window 64x8 images of V -> d_out
// ---------------------------------------------------------------------------
__global__ void cswin_rpe(const float* __restrict__ temp,
                          const float* __restrict__ cw,
                          const float* __restrict__ cb,
                          float* __restrict__ out) {
  int idx = blockIdx.x * blockDim.x + threadIdx.x;   // 8388608
  int c = idx & 127;
  int t = idx >> 7;
  int w = t & 63;
  int h = (t >> 6) & 63;
  int b = t >> 12;
  int wb = w >> 3, wc = w & 7;
  const float* vbase = temp + ((size_t)(b * 3 + 2) * 128 + c) * 4096;
  float acc = cb[c];
  #pragma unroll
  for (int dy = -1; dy <= 1; ++dy) {
    int hh = h + dy;
    if (hh < 0 || hh >= 64) continue;
    #pragma unroll
    for (int dx = -1; dx <= 1; ++dx) {
      int ww = wc + dx;                 // window-local SAME padding (64x8 image)
      if (ww < 0 || ww >= 8) continue;
      acc = fmaf(cw[c * 9 + (dy + 1) * 3 + (dx + 1)],
                 vbase[hh * 64 + wb * 8 + ww], acc);
    }
  }
  out[idx] = acc;
}

// ---------------------------------------------------------------------------
// Flash attention, one (window, head) per workgroup; 16 wave32 per block.
// Q pre-scaled by (1/sqrt(d)) * log2(e): all exponentials are plain exp2.
// 64 keys per softmax step; interleaved keys -> single b64 P store per row.
// ---------------------------------------------------------------------------
__launch_bounds__(512)
__global__ void cswin_attn(const float* __restrict__ temp,
                           float* __restrict__ out) {
  __shared__ __align__(16) unsigned short sK[CH * KSTR];        // [t_local][d]
  __shared__ __align__(16) unsigned short sV[32 * VSTR];        // [d][t_local]
  __shared__ __align__(16) unsigned short sP[16 * 16 * PSTR];   // per-wave 16x64 P

  const int bid  = blockIdx.x;        // 512 = win*4 + head
  const int win  = bid >> 2;
  const int head = bid & 3;
  const int b    = win >> 3;
  const int wb   = win & 7;

  const int tid  = threadIdx.x;
  const int wave = tid >> 5;
  const int lane = tid & 31;
  const int half = lane >> 4;
  const int l16  = lane & 15;

  // (1/sqrt(32)) * log2(e): S lands directly in log2 domain
  const float scale = 0.17677669529663687f * 1.4426950408889634f;

  const float* qg = temp + ((size_t)(b * 3 + 0) * 128 + head * 32) * 4096;
  const float* kg = temp + ((size_t)(b * 3 + 1) * 128 + head * 32) * 4096;
  const float* vg = temp + ((size_t)(b * 3 + 2) * 128 + head * 32) * 4096;
  const int spbase = wb * 8;

  // ---- Q A-fragments: direct global gather, pre-scaled, packed bf16 ----
  Frag qf[2];
  const int rowbase = wave * 32;
  #pragma unroll
  for (int rt = 0; rt < 2; ++rt) {
    int m = rowbase + rt * 16 + l16;
    int h = m >> 3, wc = m & 7;
    const float* qrow = qg + h * 64 + spbase + wc;
    #pragma unroll
    for (int j = 0; j < 8; ++j) {
      int ks = 8 * half + ((j < 4) ? (2 * j) : (16 + 2 * (j - 4)));
      qf[rt].u[j] = pk2bf(qrow[(size_t)ks * 4096] * scale,
                          qrow[(size_t)(ks + 1) * 4096] * scale);
    }
  }

  // ---- hoisted staging offsets (only + ck*1024 varies per chunk) ----
  int kOff[8], vOff[8], kLds[8], vLds[8];
  #pragma unroll
  for (int it = 0; it < 8; ++it) {
    int idx = tid + it * 512;
    int dK = idx >> 7, tlK = idx & 127;
    kOff[it] = dK * 4096 + (tlK >> 3) * 64 + spbase + (tlK & 7);
    kLds[it] = tlK * KSTR + dK;
    int tlV = idx & 127, dV = idx >> 7;
    vOff[it] = dV * 4096 + (tlV >> 3) * 64 + spbase + (tlV & 7);
    vLds[it] = dV * VSTR + tlV;
  }

  v8f   O[2][2];
  float mrow[2][8], lrow[2][8];
  v8f   zC;
  #pragma unroll
  for (int i = 0; i < 8; ++i) zC[i] = 0.0f;
  #pragma unroll
  for (int rt = 0; rt < 2; ++rt)
    #pragma unroll
    for (int r = 0; r < 8; ++r) {
      mrow[rt][r] = -3.0e38f; lrow[rt][r] = 0.0f;
      O[rt][0][r] = 0.0f;     O[rt][1][r] = 0.0f;
    }

  unsigned short* sPw = sP + wave * (16 * PSTR);

  for (int ck = 0; ck < 4; ++ck) {
    const int gsh = ck * 1024;
    __syncthreads();
    #pragma unroll
    for (int it = 0; it < 8; ++it)
      sK[kLds[it]] = t2bf(kg[kOff[it] + gsh]);     // ds_store_b16_d16_hi
    #pragma unroll
    for (int it = 0; it < 8; ++it)
      sV[vLds[it]] = t2bf(vg[vOff[it] + gsh]);
    __syncthreads();

    if (ck < 3) {   // prefetch next chunk into cache while we compute
      #pragma unroll
      for (int it = 0; it < 8; ++it) {
        __builtin_prefetch(&kg[kOff[it] + gsh + 1024], 0, 0);
        __builtin_prefetch(&vg[vOff[it] + gsh + 1024], 0, 0);
      }
    }

    #pragma unroll
    for (int js = 0; js < 2; ++js) {          // 64 keys per step
      const int kk = js * 64;
      // K B-fragments: subtile i holds tokens kk + 4*N + i (interleaved)
      Frag kf[4];
      #pragma unroll
      for (int i = 0; i < 4; ++i) {
        int tl = kk + 4 * l16 + i;
        int kb = half * 8;
        kf[i].q[0] = *(const uint4*)&sK[tl * KSTR + kb];
        kf[i].q[1] = *(const uint4*)&sK[tl * KSTR + kb + 16];
      }
      // V B-fragments: [key-half kh][dim-half dh] over token-ordered keys
      Frag vf[4];
      #pragma unroll
      for (int kh = 0; kh < 2; ++kh)
        #pragma unroll
        for (int dh = 0; dh < 2; ++dh) {
          int kb = kk + kh * 32 + half * 8;
          vf[kh * 2 + dh].q[0] = *(const uint4*)&sV[(dh * 16 + l16) * VSTR + kb];
          vf[kh * 2 + dh].q[1] = *(const uint4*)&sV[(dh * 16 + l16) * VSTR + kb + 16];
        }

      #pragma unroll
      for (int rt = 0; rt < 2; ++rt) {
        v8f s0 = __builtin_amdgcn_wmma_f32_16x16x32_bf16(
            false, qf[rt].v, false, kf[0].v, (short)0, zC, false, false);
        v8f s1 = __builtin_amdgcn_wmma_f32_16x16x32_bf16(
            false, qf[rt].v, false, kf[1].v, (short)0, zC, false, false);
        v8f s2 = __builtin_amdgcn_wmma_f32_16x16x32_bf16(
            false, qf[rt].v, false, kf[2].v, (short)0, zC, false, false);
        v8f s3 = __builtin_amdgcn_wmma_f32_16x16x32_bf16(
            false, qf[rt].v, false, kf[3].v, (short)0, zC, false, false);
        // ---- online softmax over 64 keys (log2 domain) ----
        #pragma unroll
        for (int r = 0; r < 8; ++r) {
          float a0 = s0[r], a1 = s1[r], a2 = s2[r], a3 = s3[r];
          float mx = red16_max(fmaxf(fmaxf(a0, a1), fmaxf(a2, a3)));
          float mn   = fmaxf(mrow[rt][r], mx);
          float corr = EXP2(mrow[rt][r] - mn);
          float p0 = EXP2(a0 - mn);
          float p1 = EXP2(a1 - mn);
          float p2 = EXP2(a2 - mn);
          float p3 = EXP2(a3 - mn);
          float rs = red16_sum((p0 + p1) + (p2 + p3));
          lrow[rt][r] = lrow[rt][r] * corr + rs;
          mrow[rt][r] = mn;
          O[rt][0][r] *= corr;
          O[rt][1][r] *= corr;
          // lane's 4 P values are tokens kk + 4*l16 + {0..3}: one b64 store
          int M = r + 8 * half;
          uint2 pk;
          pk.x = pk2bf(p0, p1);
          pk.y = pk2bf(p2, p3);
          *(uint2*)&sPw[M * PSTR + 4 * l16] = pk;
        }
        // ---- reload P as A fragments, 4 back-to-back PV WMMAs ----
        Frag pf0, pf1;
        pf0.q[0] = *(const uint4*)&sPw[l16 * PSTR + 8 * half];
        pf0.q[1] = *(const uint4*)&sPw[l16 * PSTR + 16 + 8 * half];
        pf1.q[0] = *(const uint4*)&sPw[l16 * PSTR + 32 + 8 * half];
        pf1.q[1] = *(const uint4*)&sPw[l16 * PSTR + 48 + 8 * half];
        O[rt][0] = __builtin_amdgcn_wmma_f32_16x16x32_bf16(
            false, pf0.v, false, vf[0].v, (short)0, O[rt][0], false, false);
        O[rt][1] = __builtin_amdgcn_wmma_f32_16x16x32_bf16(
            false, pf0.v, false, vf[1].v, (short)0, O[rt][1], false, false);
        O[rt][0] = __builtin_amdgcn_wmma_f32_16x16x32_bf16(
            false, pf1.v, false, vf[2].v, (short)0, O[rt][0], false, false);
        O[rt][1] = __builtin_amdgcn_wmma_f32_16x16x32_bf16(
            false, pf1.v, false, vf[3].v, (short)0, O[rt][1], false, false);
      }
    }
  }

  // ---- epilogue: normalize (v_rcp), add rpe (already in out), store ----
  const size_t outbase = (size_t)b * 4096 * 128;
  const int c0 = head * 32;
  #pragma unroll
  for (int rt = 0; rt < 2; ++rt) {
    #pragma unroll
    for (int r = 0; r < 8; ++r) {
      int M = rowbase + rt * 16 + r + 8 * half;
      int h = M >> 3, wc = M & 7;
      size_t o = outbase + (size_t)(h * 64 + spbase + wc) * 128 + c0;
      float inv = RCP(lrow[rt][r]);
      out[o + l16]      = fmaf(O[rt][0][r], inv, out[o + l16]);
      out[o + 16 + l16] = fmaf(O[rt][1][r], inv, out[o + 16 + l16]);
    }
  }
}

extern "C" void kernel_launch(void* const* d_in, const int* in_sizes, int n_in,
                              void* d_out, int out_size, void* d_ws, size_t ws_size,
                              hipStream_t stream) {
  const float* temp = (const float*)d_in[0];
  const float* cw   = (const float*)d_in[1];
  const float* cb   = (const float*)d_in[2];
  float* out = (float*)d_out;

  cswin_rpe<<<(16 * 64 * 64 * 128) / 256, 256, 0, stream>>>(temp, cw, cb, out);
  cswin_attn<<<512, 512, 0, stream>>>(temp, out);
}
